// GraphAttentionNetwork_84705345012449
// MI455X (gfx1250) — matrix operand order repaired and checked by reading, not compiled, not run on hardware
//
#include <hip/hip_runtime.h>
#include <hip/hip_bf16.h>
#include <math.h>

typedef __attribute__((ext_vector_type(16))) __bf16 v16bf;
typedef __attribute__((ext_vector_type(8)))  __bf16 v8bf;
typedef __attribute__((ext_vector_type(8)))  float  v8f;

// ---------------------------------------------------------------- helpers
__global__ void k_f32_to_bf16(const float* __restrict__ in, __bf16* __restrict__ out, long n) {
  long i = (long)blockIdx.x * blockDim.x + threadIdx.x;
  if (i < n) out[i] = (__bf16)in[i];
}

// out[rows_out, cols_out] <- in[rows_in, cols_in], zero-padded, f32 -> bf16 (A-side pack)
__global__ void k_pad_convert(const float* __restrict__ in, __bf16* __restrict__ out,
                              long rows_out, int cols_out, long rows_in, int cols_in) {
  long i = (long)blockIdx.x * blockDim.x + threadIdx.x;
  long n = rows_out * (long)cols_out;
  if (i >= n) return;
  long r = i / cols_out;
  int  c = (int)(i - r * cols_out);
  float v = (r < rows_in && c < cols_in) ? in[r * (long)cols_in + c] : 0.0f;
  out[i] = (__bf16)v;
}

// B-side pack: Bp[kb][n][kk] = (kb*32+kk < Kin) ? B[(kb*32+kk)*Ncol + n] : 0
// K is the padded K (multiple of 32). One lane's WMMA B-fragment becomes
// 16 contiguous bf16 (32B) at Bp + kb*Ncol*32 + col*32 + half*16.
__global__ void k_pack_B(const float* __restrict__ in, __bf16* __restrict__ out,
                         int K, int Ncol, int Kin) {
  long i = (long)blockIdx.x * blockDim.x + threadIdx.x;
  long total = (long)K * Ncol;
  if (i >= total) return;
  int blk = Ncol * 32;
  int kb = (int)(i / blk);
  int r  = (int)(i - (long)kb * blk);
  int n  = r >> 5;
  int kk = r & 31;
  int k  = kb * 32 + kk;
  float v = (k < Kin) ? in[(size_t)k * Ncol + n] : 0.0f;
  out[i] = (__bf16)v;
}

__global__ void k_copy_f32(const float* __restrict__ in, float* __restrict__ out, long n) {
  long i = (long)blockIdx.x * blockDim.x + threadIdx.x;
  if (i < n) out[i] = in[i];
}

__global__ void k_zero_f32(float* __restrict__ p, long n) {
  long i = (long)blockIdx.x * blockDim.x + threadIdx.x;
  if (i < n) p[i] = 0.0f;
}

// ------------------------------------------------- bf16 WMMA GEMM, 4 N-tiles/wave
// C[M,Ncol] = act(A[M,K] @ B[K,Ncol] + bias). B is pre-packed (k_pack_B).
// Each wave owns a 16x64 output strip: A fragment loaded once per k-step,
// reused by 4 v_wmma_f32_16x16x32_bf16.
__global__ void k_wmma_gemm4(const __bf16* __restrict__ A, const __bf16* __restrict__ Bp,
                             const float* __restrict__ bias, float* __restrict__ C,
                             __bf16* __restrict__ Cbf, int Mrows, int K, int Ncol,
                             float act_alpha /* <0 -> identity */) {
  const int NT = 4;
  int lane = threadIdx.x & 31;
  int wave = threadIdx.x >> 5;
  int groupsN = (Ncol >> 4) / NT;
  int totalGroups = (Mrows >> 4) * groupsN;
  int g = blockIdx.x * (blockDim.x >> 5) + wave;
  if (g >= totalGroups) return;
  int tm = g / groupsN;
  int gn = g - tm * groupsN;

  int half = lane >> 4;          // 0: lanes 0-15, 1: lanes 16-31
  int l15  = lane & 15;
  int arow = (tm << 4) + l15;
  int col0 = gn * (NT << 4) + l15;

  // A contiguous K layout (matches packed B): lanes 0-15 hold K 0..15, lanes 16-31 K 16..31
  const __bf16* ap = A + (size_t)arow * K + half * 16;
  // Packed B: contiguous 16 bf16 per fragment per lane
  const __bf16* bp = Bp + (size_t)col0 * 32 + half * 16;
  const size_t bstep = (size_t)Ncol * 32;   // advance one 32-deep k-block

  v8f acc[NT] = {};
  for (int k0 = 0; k0 < K; k0 += 32) {
    v8bf alo = *(const v8bf*)(ap);
    v8bf ahi = *(const v8bf*)(ap + 8);
    v16bf a;
#pragma unroll
    for (int j = 0; j < 8; ++j) { a[j] = alo[j]; a[j + 8] = ahi[j]; }

    v16bf b0 = *(const v16bf*)(bp);
    v16bf b1 = *(const v16bf*)(bp + 512);    // +16 cols * 32
    v16bf b2 = *(const v16bf*)(bp + 1024);
    v16bf b3 = *(const v16bf*)(bp + 1536);

    __builtin_prefetch(ap + 32, 0, 1);       // global_prefetch_b8 next A chunk

    acc[0] = __builtin_amdgcn_wmma_f32_16x16x32_bf16(false, a, false, b0, (short)0, acc[0], false, false);
    acc[1] = __builtin_amdgcn_wmma_f32_16x16x32_bf16(false, a, false, b1, (short)0, acc[1], false, false);
    acc[2] = __builtin_amdgcn_wmma_f32_16x16x32_bf16(false, a, false, b2, (short)0, acc[2], false, false);
    acc[3] = __builtin_amdgcn_wmma_f32_16x16x32_bf16(false, a, false, b3, (short)0, acc[3], false, false);

    ap += 32;
    bp += bstep;
  }

  // C/D layout: VGPR r -> (M = r + 8*half, N = lane&15)
  int cm = (tm << 4) + half * 8;
#pragma unroll
  for (int t = 0; t < NT; ++t) {
    int cn = col0 + (t << 4);
    float bv = bias ? bias[cn] : 0.0f;
#pragma unroll
    for (int r = 0; r < 8; ++r) {
      float v = acc[t][r] + bv;
      if (act_alpha >= 0.0f) v = (v > 0.0f) ? v : act_alpha * v;
      size_t idx = (size_t)(cm + r) * Ncol + cn;
      C[idx] = v;
      if (Cbf) Cbf[idx] = (__bf16)v;
    }
  }
}

// ------------------------------------------------- GAT edge kernels (H=3 fixed)
__global__ void k_edge_score(const float* __restrict__ hp, const float* __restrict__ a_layer,
                             const int* __restrict__ edges, float* __restrict__ s,
                             float* __restrict__ denom, int B, int N, int E, int U) {
  long idx = (long)blockIdx.x * blockDim.x + threadIdx.x;
  long total = 3L * B * E;
  if (idx >= total) return;
  int head = (int)(idx / ((long)B * E));
  long rem = idx - (long)head * B * E;
  int b = (int)(rem / E);
  int e = (int)(rem - (long)b * E);
  int src = edges[((long)head * E + e) * 2 + 0];
  int dst = edges[((long)head * E + e) * 2 + 1];
  const float* hs = hp + (((size_t)head * B + b) * N + src) * U;
  const float* hd = hp + (((size_t)head * B + b) * N + dst) * U;
  const float* al = a_layer + head * 2 * U;
  float acc = 0.0f;
  for (int u = 0; u < U; ++u) acc += hs[u] * al[u] + hd[u] * al[U + u];
  acc = (acc > 0.0f) ? acc : 0.2f * acc;                 // leaky 0.2
  acc = fminf(2.0f, fmaxf(-2.0f, acc));                  // clip
  float sv = expf(acc);
  s[idx] = sv;
  __hip_atomic_fetch_add(&denom[((long)head * B + b) * N + src], sv,
                         __ATOMIC_RELAXED, __HIP_MEMORY_SCOPE_AGENT);
}

__global__ void k_aggregate(const float* __restrict__ hp, const int* __restrict__ edges,
                            const float* __restrict__ s, const float* __restrict__ denom,
                            float* __restrict__ hout, int B, int N, int E, int U, int Dm) {
  long idx = (long)blockIdx.x * blockDim.x + threadIdx.x;
  long total = 3L * B * E;
  if (idx >= total) return;
  int head = (int)(idx / ((long)B * E));
  long rem = idx - (long)head * B * E;
  int b = (int)(rem / E);
  int e = (int)(rem - (long)b * E);
  int src = edges[((long)head * E + e) * 2 + 0];
  int dst = edges[((long)head * E + e) * 2 + 1];
  float alpha = s[idx] / denom[((long)head * B + b) * N + src];
  const float* hd = hp + (((size_t)head * B + b) * N + dst) * U;
  float* op = hout + ((size_t)b * N + src) * Dm + head * U;
  for (int u = 0; u < U; ++u)
    __hip_atomic_fetch_add(&op[u], alpha * hd[u],
                           __ATOMIC_RELAXED, __HIP_MEMORY_SCOPE_AGENT);
}

// ------------------------------------------------- tail: mean over nodes + MLP head
__global__ void k_node_mean(const float* __restrict__ h, float* __restrict__ hbar, int N, int Dm) {
  int b = blockIdx.x;
  int c = threadIdx.x;              // blockDim.x == Dm (192 = 6 waves)
  float acc = 0.0f;
  const float* p = h + ((size_t)b * N) * Dm + c;
  for (int n = 0; n < N; ++n) acc += p[(size_t)n * Dm];
  hbar[b * Dm + c] = acc / (float)N;
}

__global__ void k_head(const float* __restrict__ hbar, const float* __restrict__ pe,
                       const float* __restrict__ Wv, const float* __restrict__ W1,
                       const float* __restrict__ b1, const float* __restrict__ W2,
                       const float* __restrict__ b2, const float* __restrict__ Wout,
                       const float* __restrict__ bout, float* __restrict__ out,
                       int B, int N, int Dm, int CODE, int NL) {
  __shared__ float s_pe[16];
  __shared__ float s_ctx[32 * 128];
  __shared__ float s_z1[32 * 32];
  __shared__ float s_z2[32 * 32];
  __shared__ float s_o[32];
  int tid = threadIdx.x;            // blockDim.x == 1024
  if (tid < CODE) {
    float acc = 0.0f;
    for (int n = 0; n < N; ++n) acc += pe[(size_t)n * CODE + tid];
    s_pe[tid] = acc / (float)N;
  }
  __syncthreads();
  for (int i = tid; i < B * 128; i += blockDim.x) {   // ctx = [hbar | pebar] @ Wv
    int b = i >> 7, j = i & 127;
    float acc = 0.0f;
    const float* hb = hbar + b * Dm;
    for (int c = 0; c < Dm; ++c) acc += hb[c] * Wv[(size_t)c * 128 + j];
    for (int c = 0; c < CODE; ++c) acc += s_pe[c] * Wv[(size_t)(Dm + c) * 128 + j];
    s_ctx[i] = acc;
  }
  __syncthreads();
  if (tid < B * 32) {
    int b = tid >> 5, j = tid & 31;
    float acc = b1[j];
    for (int c = 0; c < 128; ++c) acc += s_ctx[b * 128 + c] * W1[c * 32 + j];
    s_z1[tid] = (acc > 0.0f) ? acc : 0.3f * acc;
  }
  __syncthreads();
  if (tid < B * 32) {
    int b = tid >> 5, j = tid & 31;
    float acc = b2[j];
    for (int c = 0; c < 32; ++c) acc += s_z1[b * 32 + c] * W2[c * 32 + j];
    s_z2[tid] = (acc > 0.0f) ? acc : 0.3f * acc;
  }
  __syncthreads();
  if (tid < B) {
    float acc = bout[0];
    for (int c = 0; c < 32; ++c) acc += s_z2[tid * 32 + c] * Wout[c];
    s_o[tid] = tanhf(acc);
  }
  __syncthreads();
  long total = (long)B * NL;
  for (long i = tid; i < total; i += blockDim.x) out[i] = s_o[i / NL];
}

// ---------------------------------------------------------------- launcher
extern "C" void kernel_launch(void* const* d_in, const int* in_sizes, int n_in,
                              void* d_out, int out_size, void* d_ws, size_t ws_size,
                              hipStream_t stream) {
  const float* x       = (const float*)d_in[0];
  const float* W_pre   = (const float*)d_in[1];
  const float* b_pre   = (const float*)d_in[2];
  const float* W_gat   = (const float*)d_in[3];
  const float* a_gat   = (const float*)d_in[4];
  const float* node_pe = (const float*)d_in[5];
  const float* Wv      = (const float*)d_in[6];
  const float* W1      = (const float*)d_in[7];
  const float* b1      = (const float*)d_in[8];
  const float* W2      = (const float*)d_in[9];
  const float* b2      = (const float*)d_in[10];
  const float* Wout    = (const float*)d_in[11];
  const float* bout    = (const float*)d_in[12];
  const int*   edges   = (const int*)d_in[13];
  float* out = (float*)d_out;
  (void)n_in; (void)ws_size;

  const int B = 32, N = 2048, F = 16, U = 64, H = 3, L = 3, Dm = 192, CODE = 16;
  const int E  = in_sizes[13] / (H * 2);     // 8192
  const int NL = out_size / B;               // 3072
  const size_t M = (size_t)B * N;            // 65536 node-rows

  // ---- workspace carve (256B aligned slabs) ----
  char* w = (char*)d_ws;
  auto carve = [&](size_t bytes) { char* p = w; w += (bytes + 255) & ~(size_t)255; return p; };
  __bf16* x_bf    = (__bf16*)carve(M * 32 * 2);                 // K padded 16->32
  __bf16* Wpre_pk = (__bf16*)carve((size_t)32 * Dm * 2);        // packed, K padded 16->32
  __bf16* Wgat_pk = (__bf16*)carve((size_t)L * H * Dm * U * 2); // packed per head
  float*  h_a     = (float*)carve(M * Dm * 4);
  float*  h_b     = (float*)carve(M * Dm * 4);
  __bf16* h_bf    = (__bf16*)carve(M * Dm * 2);
  float*  hp      = (float*)carve((size_t)H * M * U * 4);
  float*  s_buf   = (float*)carve((size_t)H * B * E * 4);
  float*  denom   = (float*)carve((size_t)H * B * N * 4);
  float*  hbar    = (float*)carve((size_t)B * Dm * 4);

  auto cdiv = [](long a, long b) { return (a + b - 1) / b; };

  // ---- pack operands to bf16 ----
  k_pad_convert<<<cdiv((long)M * 32, 256), 256, 0, stream>>>(x, x_bf, (long)M, 32, (long)M, F);
  k_pack_B<<<cdiv(32L * Dm, 256), 256, 0, stream>>>(W_pre, Wpre_pk, 32, Dm, F);
  for (int m = 0; m < L * H; ++m)
    k_pack_B<<<cdiv((long)Dm * U, 256), 256, 0, stream>>>(
        W_gat + (size_t)m * Dm * U, Wgat_pk + (size_t)m * Dm * U, Dm, U, Dm);

  // ---- preprocess: h = LeakyReLU_0.3(x @ W_pre + b_pre), f32 + bf16 outputs ----
  {
    long groups = (long)(M / 16) * ((Dm / 16) / 4);
    k_wmma_gemm4<<<cdiv(groups, 8), 256, 0, stream>>>(x_bf, Wpre_pk, b_pre, h_a, h_bf,
                                                      (int)M, 32, Dm, 0.3f);
  }

  // ---- 3 GAT layers ----
  float* h_cur = h_a;
  float* h_next = h_b;
  for (int l = 0; l < L; ++l) {
    k_zero_f32<<<cdiv((long)H * B * N, 256), 256, 0, stream>>>(denom, (long)H * B * N);
    k_copy_f32<<<cdiv((long)M * Dm, 256), 256, 0, stream>>>(h_cur, h_next, (long)M * Dm); // residual init
    for (int i = 0; i < H; ++i) {
      long groups = (long)(M / 16) * ((U / 16) / 4);
      k_wmma_gemm4<<<cdiv(groups, 8), 256, 0, stream>>>(
          h_bf, Wgat_pk + (size_t)(l * H + i) * Dm * U, nullptr,
          hp + (size_t)i * M * U, nullptr, (int)M, Dm, U, -1.0f);
    }
    long et = (long)H * B * E;
    k_edge_score<<<cdiv(et, 256), 256, 0, stream>>>(hp, a_gat + (size_t)l * H * 2 * U,
                                                    edges, s_buf, denom, B, N, E, U);
    k_aggregate<<<cdiv(et, 256), 256, 0, stream>>>(hp, edges, s_buf, denom, h_next,
                                                   B, N, E, U, Dm);
    float* t = h_cur; h_cur = h_next; h_next = t;
    if (l < L - 1)
      k_f32_to_bf16<<<cdiv((long)M * Dm, 256), 256, 0, stream>>>(h_cur, h_bf, (long)M * Dm);
  }

  // ---- tail: mean over nodes (linearity of mean makes V-GEMM exact at [B,208]) ----
  k_node_mean<<<B, Dm, 0, stream>>>(h_cur, hbar, N, Dm);
  k_head<<<1, 1024, 0, stream>>>(hbar, node_pe, Wv, W1, b1, W2, b2, Wout, bout, out,
                                 B, N, Dm, CODE, NL);
}
